// SparseMoEBlock_17368847745257
// MI455X (gfx1250) — matrix operand order repaired and checked by reading, not compiled, not run on hardware
//
#include <hip/hip_runtime.h>
#include <stdint.h>

#define NTOK 8192
#define DM   512
#define DFF  1024
#define NEXP 8

typedef __attribute__((ext_vector_type(16))) __bf16          bf16x16;
typedef __attribute__((ext_vector_type(8)))  float           f32x8;
typedef __attribute__((ext_vector_type(16))) unsigned short  u16x16;

union FragCast { u16x16 u; bf16x16 b; };

static __device__ __forceinline__ unsigned short f2bf(float f) {
  union { float f; unsigned u; } v; v.f = f;
  unsigned r = v.u + 0x7FFFu + ((v.u >> 16) & 1u);   // round-to-nearest-even
  return (unsigned short)(r >> 16);
}

static __device__ __forceinline__ float fast_silu(float v) {
  // v * sigmoid(v) with v_rcp_f32 instead of IEEE divide
  return v * __builtin_amdgcn_rcpf(1.f + __expf(-v));
}

// CDNA5 async memory->LDS staging (ASYNCcnt-tracked), GVS addressing:
//   LDS target comes from the VDST VGPR, global source is SGPR base plus VGPR byte offset
#define ASYNC_LDS_B64(ldsa, voff, sbase)                                     \
  asm volatile("global_load_async_to_lds_b64 %0, %1, %2 offset:0"            \
               :: "v"(ldsa), "v"(voff), "s"(sbase) : "memory")
#define ASYNC_LDS_B128x2(ldsa, voff, sbase)                                  \
  asm volatile("global_load_async_to_lds_b128 %0, %1, %2 offset:0\n\t"       \
               "global_load_async_to_lds_b128 %0, %1, %2 offset:16"          \
               :: "v"(ldsa), "v"(voff), "s"(sbase) : "memory")
// async loads complete in order: <=3 outstanding keeps the newest 3-instr group
// in flight while guaranteeing the previous group has landed in LDS
#define WAIT_ASYNC3() asm volatile("s_wait_asynccnt 3" ::: "memory")
#define WAIT_ASYNC0() asm volatile("s_wait_asynccnt 0" ::: "memory")

// One K-step: load A/B fragments with k = 16*h + q for both A and B and
// run two 16x16x32 bf16 WMMAs (two adjacent N sub-tiles share the A fragment).
static __device__ __forceinline__ void frag_mma(
    const unsigned short* __restrict__ cA, const unsigned short* __restrict__ cB,
    int arow, int brow, int h, f32x8& acc0, f32x8& acc1)
{
  FragCast a;
  #pragma unroll
  for (int q = 0; q < 16; ++q) {
    int kq = h * 16 + q;
    a.u[q] = cA[arow * 40 + kq];
  }
  FragCast b0, b1;
  #pragma unroll
  for (int q = 0; q < 16; ++q) b0.u[q] = cB[brow * 40 + h * 16 + q];
  #pragma unroll
  for (int q = 0; q < 16; ++q) b1.u[q] = cB[(brow + 16) * 40 + h * 16 + q];
  acc0 = __builtin_amdgcn_wmma_f32_16x16x32_bf16(false, a.b, false, b0.b, (short)0, acc0, false, false);
  acc1 = __builtin_amdgcn_wmma_f32_16x16x32_bf16(false, a.b, false, b1.b, (short)0, acc1, false, false);
}

// ---------------- Router: one wave32 per token ----------------
__global__ __launch_bounds__(256) void router_kernel(
    const float* __restrict__ x, const float* __restrict__ wr,
    int* __restrict__ topE, float* __restrict__ topG, int* __restrict__ cnt)
{
  const int lane = threadIdx.x & 31;
  const int t = blockIdx.x * 8 + (threadIdx.x >> 5);
  float acc[NEXP];
  #pragma unroll
  for (int e = 0; e < NEXP; ++e) acc[e] = 0.f;
  const float* xr = x + (size_t)t * DM;
  for (int k = lane; k < DM; k += 32) {
    float xv = xr[k];
    #pragma unroll
    for (int e = 0; e < NEXP; ++e) acc[e] += xv * wr[e * DM + k];
  }
  #pragma unroll
  for (int e = 0; e < NEXP; ++e) {
    #pragma unroll
    for (int off = 16; off > 0; off >>= 1)
      acc[e] += __shfl_xor(acc[e], off, 32);
  }
  if (lane == 0) {
    int i0 = 0;
    #pragma unroll
    for (int e = 1; e < NEXP; ++e) if (acc[e] > acc[i0]) i0 = e;     // first max (jax tie rule)
    int i1 = (i0 == 0) ? 1 : 0;
    #pragma unroll
    for (int e = 0; e < NEXP; ++e) if (e != i0 && acc[e] > acc[i1]) i1 = e;
    // renormalized softmax over top-2 == sigmoid of logit difference
    float g0 = 1.f / (1.f + __expf(acc[i1] - acc[i0]));
    float g1 = 1.f - g0;
    topE[t * 2]     = i0; topE[t * 2 + 1] = i1;
    topG[t * 2]     = g0; topG[t * 2 + 1] = g1;
    atomicAdd(&cnt[i0], 1);
    atomicAdd(&cnt[i1], 1);
  }
}

// ---------------- Tiny exclusive scan over 8 expert counts ----------------
__global__ void scan_kernel(const int* __restrict__ cnt, int* __restrict__ offs) {
  if (threadIdx.x == 0) {
    int s = 0;
    for (int e = 0; e < NEXP; ++e) { offs[e] = s; s += cnt[e]; }
    offs[NEXP] = s;   // == 2*NTOK
  }
}

// ---------------- Compact tokens into per-expert slot lists ----------------
__global__ __launch_bounds__(256) void build_kernel(
    const int* __restrict__ topE, const float* __restrict__ topG,
    const int* __restrict__ offs, int* __restrict__ cursor,
    int* __restrict__ tok, float* __restrict__ agate)
{
  int t = blockIdx.x * blockDim.x + threadIdx.x;
  if (t >= NTOK) return;
  #pragma unroll
  for (int j = 0; j < 2; ++j) {
    int e = topE[t * 2 + j];
    int pos = atomicAdd(&cursor[e], 1);
    int slot = offs[e] + pos;
    tok[slot]   = t;
    agate[slot] = topG[t * 2 + j];
  }
}

// ---------------- fp32 -> bf16 conversion (2 elems/thread, packed dword store) ----------------
__global__ __launch_bounds__(256) void cvt2_kernel(
    const float2* __restrict__ s, unsigned* __restrict__ d, int n2)
{
  int i = blockIdx.x * blockDim.x + threadIdx.x;
  if (i < n2) {
    float2 v = s[i];
    d[i] = (unsigned)f2bf(v.x) | ((unsigned)f2bf(v.y) << 16);
  }
}

// ---------------- GEMM1: H = silu(Xg @ W1[e]^T), 32x128 tile / block ----------------
// Triple-buffered async-to-LDS pipeline (prefetch distance 2), K-step 32,
// last two iterations peeled so the hot loop has one unconditional wait.
__global__ __launch_bounds__(256) void gemm1_kernel(
    const unsigned short* __restrict__ xb, const unsigned short* __restrict__ w1b,
    const int* __restrict__ offs, const int* __restrict__ tok,
    unsigned short* __restrict__ hb)
{
  const int e     = blockIdx.z;
  const int begin = offs[e], end = offs[e + 1];
  const int m0    = blockIdx.y * 32;
  if (m0 >= end - begin) return;
  const int n0 = blockIdx.x * 128;

  __shared__ int            sTok[32];
  __shared__ unsigned short sA[3][32 * 40];    // 32 rows x 32 k, stride 40 halfs
  __shared__ unsigned short sB[3][128 * 40];   // 128 n  x 32 k

  const int tid = threadIdx.x;
  if (tid < 32) {
    int slot = begin + m0 + tid;
    if (slot >= end) slot = end - 1;           // clamp (padding rows; stores guarded)
    sTok[tid] = tok[slot];
  }
  __syncthreads();

  // staging coordinates (per thread): A -> one b64, B -> two b128 (one 32B row chunk)
  const int ar = tid >> 3, ak = (tid & 7) * 4;   // A: row 0..31, half offset 0..28
  const int br = tid >> 1, bk = (tid & 1) * 16;  // B: row 0..127, half offset 0/16
  const int tokR = sTok[ar];
  unsigned aOff = ((unsigned)tokR * DM + (unsigned)ak) * 2u;   // byte offset into xb
  unsigned bOff = ((unsigned)br * DM + (unsigned)bk) * 2u;     // byte offset from wbase
  const unsigned short* wbase = w1b + (size_t)e * DFF * DM + (size_t)n0 * DM;

  unsigned ldsA[3], ldsB[3];
  #pragma unroll
  for (int b = 0; b < 3; ++b) {
    ldsA[b] = (unsigned)(uintptr_t)&sA[b][ar * 40 + ak];
    ldsB[b] = (unsigned)(uintptr_t)&sB[b][br * 40 + bk];
  }

  const int NSTEP = DM / 32;                    // 16
  // prologue: stage steps 0,1 into buffers 0,1
  ASYNC_LDS_B64(ldsA[0], aOff, xb);
  ASYNC_LDS_B128x2(ldsB[0], bOff, wbase);
  ASYNC_LDS_B64(ldsA[1], aOff + 64u, xb);
  ASYNC_LDS_B128x2(ldsB[1], bOff + 64u, wbase);
  WAIT_ASYNC3();                                // buffer 0 landed
  __syncthreads();

  const int lane = tid & 31, wv = tid >> 5;
  const int wm = wv & 1, wn = wv >> 1;          // 2x4 wave grid
  const int h = lane >> 4, l16 = lane & 15;
  const int arow = wm * 16 + l16;
  const int brow = wn * 32 + l16;
  f32x8 acc0 = {}, acc1 = {};

  int cur = 0;
  for (int s = 0; s < NSTEP - 2; ++s) {         // steady state: always prefetch + WAIT3
    int pf = cur + 2; if (pf >= 3) pf -= 3;
    unsigned aO = aOff + (unsigned)(s + 2) * 64u;
    unsigned bO = bOff + (unsigned)(s + 2) * 64u;
    ASYNC_LDS_B64(ldsA[pf], aO, xb);
    ASYNC_LDS_B128x2(ldsB[pf], bO, wbase);
    frag_mma(sA[cur], sB[cur], arow, brow, h, acc0, acc1);
    WAIT_ASYNC3();                              // slab s+1 guaranteed in LDS
    __syncthreads();                            // all waves done reading slab s
    cur = (cur == 2) ? 0 : cur + 1;
  }
  // peeled step NSTEP-2: nothing left to issue, drain the final slab
  frag_mma(sA[cur], sB[cur], arow, brow, h, acc0, acc1);
  WAIT_ASYNC0();
  __syncthreads();
  cur = (cur == 2) ? 0 : cur + 1;
  // peeled step NSTEP-1: last slab; no wait/barrier needed afterwards
  frag_mma(sA[cur], sB[cur], arow, brow, h, acc0, acc1);

  #pragma unroll
  for (int r = 0; r < 8; ++r) {                 // C/D: m = r + 8*(lane/16), n = lane%16
    int mrow = wm * 16 + r + 8 * h;
    int slot = begin + m0 + mrow;
    if (slot < end) {
      int ng = n0 + wn * 32 + l16;
      hb[(size_t)slot * DFF + ng]      = f2bf(fast_silu(acc0[r]));
      hb[(size_t)slot * DFF + ng + 16] = f2bf(fast_silu(acc1[r]));
    }
  }
}

// ---------------- GEMM2: out[t] += gate * (H @ W2[e]^T), atomic scatter ----------------
__global__ __launch_bounds__(256) void gemm2_kernel(
    const unsigned short* __restrict__ hb, const unsigned short* __restrict__ w2b,
    const int* __restrict__ offs, const int* __restrict__ tok,
    const float* __restrict__ agate, float* __restrict__ out)
{
  const int e     = blockIdx.z;
  const int begin = offs[e], end = offs[e + 1];
  const int m0    = blockIdx.y * 32;
  if (m0 >= end - begin) return;
  const int n0 = blockIdx.x * 128;

  __shared__ int            sTok[32];
  __shared__ float          sG[32];
  __shared__ unsigned short sA[3][32 * 40];
  __shared__ unsigned short sB[3][128 * 40];

  const int tid = threadIdx.x;
  if (tid < 32) {
    int slot = begin + m0 + tid;
    if (slot >= end) slot = end - 1;
    sTok[tid] = tok[slot];
    sG[tid]   = agate[slot];
  }
  __syncthreads();

  const int ar = tid >> 3, ak = (tid & 7) * 4;
  const int br = tid >> 1, bk = (tid & 1) * 16;
  int aSlot = begin + m0 + ar; if (aSlot >= end) aSlot = end - 1;   // H rows contiguous
  unsigned aOff = ((unsigned)aSlot * DFF + (unsigned)ak) * 2u;      // byte offset into hb
  unsigned bOff = ((unsigned)br * DFF + (unsigned)bk) * 2u;
  const unsigned short* wbase = w2b + (size_t)e * DM * DFF + (size_t)n0 * DFF;

  unsigned ldsA[3], ldsB[3];
  #pragma unroll
  for (int b = 0; b < 3; ++b) {
    ldsA[b] = (unsigned)(uintptr_t)&sA[b][ar * 40 + ak];
    ldsB[b] = (unsigned)(uintptr_t)&sB[b][br * 40 + bk];
  }

  const int NSTEP = DFF / 32;                   // 32
  ASYNC_LDS_B64(ldsA[0], aOff, hb);
  ASYNC_LDS_B128x2(ldsB[0], bOff, wbase);
  ASYNC_LDS_B64(ldsA[1], aOff + 64u, hb);
  ASYNC_LDS_B128x2(ldsB[1], bOff + 64u, wbase);
  WAIT_ASYNC3();
  __syncthreads();

  const int lane = tid & 31, wv = tid >> 5;
  const int wm = wv & 1, wn = wv >> 1;
  const int h = lane >> 4, l16 = lane & 15;
  const int arow = wm * 16 + l16;
  const int brow = wn * 32 + l16;
  f32x8 acc0 = {}, acc1 = {};

  int cur = 0;
  for (int s = 0; s < NSTEP - 2; ++s) {
    int pf = cur + 2; if (pf >= 3) pf -= 3;
    unsigned aO = aOff + (unsigned)(s + 2) * 64u;
    unsigned bO = bOff + (unsigned)(s + 2) * 64u;
    ASYNC_LDS_B64(ldsA[pf], aO, hb);
    ASYNC_LDS_B128x2(ldsB[pf], bO, wbase);
    frag_mma(sA[cur], sB[cur], arow, brow, h, acc0, acc1);
    WAIT_ASYNC3();
    __syncthreads();
    cur = (cur == 2) ? 0 : cur + 1;
  }
  frag_mma(sA[cur], sB[cur], arow, brow, h, acc0, acc1);
  WAIT_ASYNC0();
  __syncthreads();
  cur = (cur == 2) ? 0 : cur + 1;
  frag_mma(sA[cur], sB[cur], arow, brow, h, acc0, acc1);

  #pragma unroll
  for (int r = 0; r < 8; ++r) {
    int mrow = wm * 16 + r + 8 * h;
    int slot = begin + m0 + mrow;
    if (slot < end) {
      int   tk = sTok[mrow];
      float g  = sG[mrow];
      int   ng = n0 + wn * 32 + l16;
      // exactly 2 contributions per out element; float add is commutative -> deterministic
      atomicAdd(&out[(size_t)tk * DM + ng],      g * acc0[r]);
      atomicAdd(&out[(size_t)tk * DM + ng + 16], g * acc1[r]);
    }
  }
}

extern "C" void kernel_launch(void* const* d_in, const int* in_sizes, int n_in,
                              void* d_out, int out_size, void* d_ws, size_t ws_size,
                              hipStream_t stream)
{
  const float* x  = (const float*)d_in[0];
  const float* wr = (const float*)d_in[1];
  const float* w1 = (const float*)d_in[2];
  const float* w2 = (const float*)d_in[3];
  float* out = (float*)d_out;

  char* ws = (char*)d_ws;
  size_t o = 0;
  auto alloc = [&](size_t bytes) -> void* {
    void* p = ws + o;
    o += (bytes + 255) & ~(size_t)255;
    return p;
  };
  int*            cnt    = (int*)  alloc(NEXP * 4);
  int*            offs   = (int*)  alloc((NEXP + 1) * 4);
  int*            cursor = (int*)  alloc(NEXP * 4);
  int*            topE   = (int*)  alloc((size_t)NTOK * 2 * 4);
  float*          topG   = (float*)alloc((size_t)NTOK * 2 * 4);
  int*            tok    = (int*)  alloc((size_t)NTOK * 2 * 4);
  float*          agate  = (float*)alloc((size_t)NTOK * 2 * 4);
  unsigned short* xb     = (unsigned short*)alloc((size_t)NTOK * DM * 2);
  unsigned short* w1b    = (unsigned short*)alloc((size_t)NEXP * DFF * DM * 2);
  unsigned short* w2b    = (unsigned short*)alloc((size_t)NEXP * DM * DFF * 2);
  unsigned short* hb     = (unsigned short*)alloc((size_t)NTOK * 2 * DFF * 2);

  hipMemsetAsync(cnt,    0, NEXP * 4, stream);
  hipMemsetAsync(cursor, 0, NEXP * 4, stream);
  hipMemsetAsync(out,    0, (size_t)out_size * sizeof(float), stream);

  router_kernel<<<NTOK / 8, 256, 0, stream>>>(x, wr, topE, topG, cnt);
  scan_kernel<<<1, 32, 0, stream>>>(cnt, offs);
  build_kernel<<<NTOK / 256, 256, 0, stream>>>(topE, topG, offs, cursor, tok, agate);

  cvt2_kernel<<<(NTOK * DM / 2) / 256, 256, 0, stream>>>((const float2*)x, (unsigned*)xb, NTOK * DM / 2);
  cvt2_kernel<<<(NEXP * DFF * DM / 2) / 256, 256, 0, stream>>>((const float2*)w1, (unsigned*)w1b, NEXP * DFF * DM / 2);
  cvt2_kernel<<<(NEXP * DM * DFF / 2) / 256, 256, 0, stream>>>((const float2*)w2, (unsigned*)w2b, NEXP * DM * DFF / 2);

  dim3 g1(DFF / 128, NTOK / 32, NEXP);   // (8, 256, 8); blocks past an expert's count exit early
  gemm1_kernel<<<g1, 256, 0, stream>>>(xb, w1b, offs, tok, hb);
  dim3 g2(DM / 128, NTOK / 32, NEXP);    // (4, 256, 8)
  gemm2_kernel<<<g2, 256, 0, stream>>>(hb, w2b, offs, tok, agate, out);
}
